// TransformeCostVolume_58119497450280
// MI455X (gfx1250) — matrix operand order repaired and checked by reading, not compiled, not run on hardware
//
#include <hip/hip_runtime.h>
#include <hip/hip_bf16.h>

typedef __bf16 bf16;
typedef __attribute__((ext_vector_type(16))) __bf16 v16bf;
typedef __attribute__((ext_vector_type(8)))  __bf16 v8bf;
typedef __attribute__((ext_vector_type(8)))  float  v8f;
typedef __attribute__((ext_vector_type(4)))  unsigned int v4u;
typedef __attribute__((ext_vector_type(8)))  int v8i;
typedef __attribute__((ext_vector_type(4)))  int v4i;

#define NLAYERS 6

#if defined(__has_builtin)
#if __has_builtin(__builtin_amdgcn_tensor_load_to_lds) && __has_builtin(__builtin_amdgcn_s_wait_tensorcnt)
#define HAVE_TDM 1
#endif
#endif

// ---------------------------------------------------------------------------
// WMMA operand loaders (wave32, 16x16x32 bf16).
// A (16xK, row-major, per-lane row = lane&15):
//   elements 0..7  hold K = half*8 + 0..7
//   elements 8..15 hold K = 16 + half*8 + 0..7       (half = lane>>4)
// B (Kx16, stored as N x K row-major, per-lane row = lane&15 = N):
//   elements 0..7 hold K = half*8 + 0..7, elements 8..15 hold K = 16 + half*8 + 0..7
// C/D v8f: element r -> M = r + half*8, N = lane&15
// ---------------------------------------------------------------------------
struct V16U { union { v16bf v; v8bf h[2]; }; };

__device__ inline v16bf ld_a_full(const bf16* row, int half, int koff) {
  V16U u;
  u.h[0] = *(const v8bf*)(row + koff + half * 8);
  u.h[1] = *(const v8bf*)(row + koff + 16 + half * 8);
  return u.v;
}
__device__ inline v16bf ld_b_full(const bf16* row, int half, int koff) {
  V16U u;
  const bf16* p = row + koff + half * 8;
  u.h[0] = *(const v8bf*)(p);
  u.h[1] = *(const v8bf*)(p + 16);
  return u.v;
}
// K=16 variants (upper K zero-padded)
__device__ inline v16bf ld_a_k16(const bf16* row, int half) {
  V16U u;
  u.h[0] = *(const v8bf*)(row + half * 8);
  #pragma unroll
  for (int i = 0; i < 8; ++i) u.h[1][i] = (bf16)0.0f;
  return u.v;
}
__device__ inline v16bf ld_b_k16(const bf16* row, int half) {
  V16U u;
  v8bf lo = *(const v8bf*)(row + half * 8);
  v8bf hi;
  #pragma unroll
  for (int i = 0; i < 8; ++i) hi[i] = (bf16)0.0f;
  u.h[0] = lo; u.h[1] = hi;
  return u.v;
}
__device__ inline v8f wmma_bf16(v16bf a, v16bf b, v8f c) {
  return __builtin_amdgcn_wmma_f32_16x16x32_bf16(false, a, false, b,
                                                 (short)0, c, false, false);
}

#ifdef HAVE_TDM
// TDM: DMA a 64x128 bf16 tile (row stride 128 elems) from global into LDS.
// D# built per CDNA5 ISA sec 8: group0 = {count|..., lds_addr, ga_lo, ga_hi|type},
// group1 = flags/dims/strides; groups 2,3 zero (2D tensor, tile_dim2=0).
__device__ inline void tdm_load_w64x128(const bf16* gsrc, unsigned lds_off) {
  unsigned long long ga = (unsigned long long)(uintptr_t)gsrc;
  v4u g0;
  g0[0] = 1u;                                   // count=1, is_restore=0, no gather
  g0[1] = lds_off;                              // LDS byte address
  g0[2] = (unsigned)(ga & 0xFFFFFFFFu);         // global_addr[31:0]
  g0[3] = (unsigned)((ga >> 32) & 0x1FFFFFFu)   // global_addr[56:32]
          | (2u << 30);                         // type=2 ("image")
  v8i g1;
  g1[0] = 0x00010000;                           // data_size=1 (2 bytes), wg_mask=0
  g1[1] = (128 << 16);                          // tensor_dim0[15:0]=128 (bits 63:48)
  g1[2] = (64  << 16);                          // tensor_dim1[15:0]=64  (bits 95:80)
  g1[3] = (128 << 16);                          // tile_dim0=128 (bits 127:112)
  g1[4] = 64;                                   // tile_dim1=64, tile_dim2=0
  g1[5] = 128;                                  // tensor_dim0_stride[31:0]=128
  g1[6] = 0;                                    // stride hi / dim1_stride lo
  g1[7] = 0;
  v4i z4; z4[0] = z4[1] = z4[2] = z4[3] = 0;
#if __clang_major__ >= 23
  v8i z8;
  #pragma unroll
  for (int i = 0; i < 8; ++i) z8[i] = 0;
  __builtin_amdgcn_tensor_load_to_lds(g0, g1, z4, z4, z8, 0);
#else
  __builtin_amdgcn_tensor_load_to_lds(g0, g1, z4, z4, 0);
#endif
}
#endif

// ---------------------------------------------------------------------------
// Elementwise / setup kernels
// ---------------------------------------------------------------------------
__global__ __launch_bounds__(256)
void k_cvt_bf16(const float* __restrict__ s, bf16* __restrict__ d, int n) {
  int i = blockIdx.x * 256 + threadIdx.x;
  if (i < n) d[i] = (bf16)s[i];
}

// feat[w][n][c]: n<64 -> feat_left[0][c][n][w] ; else feat_right[0][c][n-64][w]
__global__ __launch_bounds__(256)
void k_build_feat(const float* __restrict__ fl, const float* __restrict__ fr,
                  float* __restrict__ feat) {
  int idx = blockIdx.x * 256 + threadIdx.x;
  if (idx >= 128 * 128 * 128) return;
  int c = idx & 127, n = (idx >> 7) & 127, w = idx >> 14;
  float v;
  if (n < 64) v = fl[(size_t)(c * 64 + n) * 128 + w];
  else        v = fr[(size_t)(c * 64 + (n - 64)) * 128 + w];
  feat[idx] = v;
}

// pos_sel[i][j][c] = pos_enc[127 - i + j][c]  (bf16)
__global__ __launch_bounds__(256)
void k_build_pos(const float* __restrict__ pe, bf16* __restrict__ ps) {
  int idx = blockIdx.x * 256 + threadIdx.x;
  if (idx >= 128 * 128 * 128) return;
  int c = idx & 127, j = (idx >> 7) & 127, i = idx >> 14;
  ps[idx] = (bf16)pe[(size_t)(127 - i + j) * 128 + c];
}

// One wave per row of 128; row -> (w, nn) via n_shift; src slice at n_off.
__global__ __launch_bounds__(256)
void k_layernorm(const float* __restrict__ feat, const float* __restrict__ g,
                 const float* __restrict__ b, bf16* __restrict__ out,
                 int rows, int n_shift, int n_off) {
  int row = blockIdx.x * 8 + (threadIdx.x >> 5);
  if (row >= rows) return;
  int lane = threadIdx.x & 31;
  int nn = row & ((1 << n_shift) - 1), w = row >> n_shift;
  const float* src = feat + (size_t)((w << 7) + n_off + nn) * 128;
  float4 x = *((const float4*)src + lane);
  float s  = x.x + x.y + x.z + x.w;
  float sq = x.x * x.x + x.y * x.y + x.z * x.z + x.w * x.w;
  for (int d = 16; d; d >>= 1) { s += __shfl_xor(s, d, 32); sq += __shfl_xor(sq, d, 32); }
  float m  = s * (1.0f / 128.0f);
  float vr = sq * (1.0f / 128.0f) - m * m;
  float rs = rsqrtf(vr + 1e-5f);
  float4 g4 = *((const float4*)g + lane);
  float4 b4 = *((const float4*)b + lane);
  bf16* dst = out + (size_t)row * 128 + lane * 4;
  dst[0] = (bf16)((x.x - m) * rs * g4.x + b4.x);
  dst[1] = (bf16)((x.y - m) * rs * g4.y + b4.y);
  dst[2] = (bf16)((x.z - m) * rs * g4.z + b4.z);
  dst[3] = (bf16)((x.w - m) * rs * g4.w + b4.w);
}

// ---------------------------------------------------------------------------
// Projection GEMM: C[M,N] = A[M,128] @ W[N,128]^T + bias.
// Block = 8 waves; block handles 8 consecutive M-tiles x one 64-wide N group.
// The 64x128 bf16 W tile (16 KB) is staged into LDS by the Tensor Data Mover
// (wave 0 issues tensor_load_to_lds, waits TENSORcnt, then block barrier).
// Each wave: 16x64 output = 4 accumulators, 16 WMMAs; per K-step all four B
// fragments are loaded first, then the 4 WMMAs issue back-to-back.
// mode 0: self QKV (N=384): q(x0.25)->p0, k->p1, v->p2 as V^T[nn,e,c,w]
// mode 1: cross KV  (N=256): k->p0, v->p2 as V^T
// mode 2: cross Q   (N=128): q(x0.25)->p0
// mode 3: pos       (N=256): q_r(x0.25)->p0, k_r->p1
// mode 4: out-proj  (N=128): feat[(w*128)+n_off+nn][col] += val
// ---------------------------------------------------------------------------
__global__ __launch_bounds__(256)
void k_gemm_proj(const bf16* __restrict__ A, const bf16* __restrict__ Wm,
                 const float* __restrict__ bias, int M, int N,
                 int mode, int n_shift, int n_off, float qscale,
                 bf16* __restrict__ p0, bf16* __restrict__ p1,
                 bf16* __restrict__ p2, float* __restrict__ featp) {
  __shared__ __align__(16) bf16 smemW[64 * 128];
  int tiles_n4 = N >> 6;
  int tn4 = blockIdx.x % tiles_n4;
  int tm  = (blockIdx.x / tiles_n4) * 8 + (threadIdx.x >> 5);
  int lane = threadIdx.x & 31;
  const bf16* wsrc = Wm + (size_t)tn4 * 64 * 128;
#ifdef HAVE_TDM
  if ((threadIdx.x >> 5) == 0) {
    tdm_load_w64x128(wsrc, (unsigned)(unsigned long long)(uintptr_t)smemW);
    __builtin_amdgcn_s_wait_tensorcnt(0);
  }
#else
  for (int t = threadIdx.x; t < 1024; t += 256)
    ((v8bf*)smemW)[t] = ((const v8bf*)wsrc)[t];
#endif
  __syncthreads();

  int half = lane >> 4, l15 = lane & 15;
  const bf16* arow = A + (size_t)(tm * 16 + l15) * 128;
  v8f acc[4];
  #pragma unroll
  for (int j = 0; j < 4; ++j)
    #pragma unroll
    for (int i = 0; i < 8; ++i) acc[j][i] = 0.0f;
  #pragma unroll
  for (int kk = 0; kk < 4; ++kk) {
    v16bf a = ld_a_full(arow, half, kk * 32);
    v16bf bfrag[4];
    #pragma unroll
    for (int j = 0; j < 4; ++j)
      bfrag[j] = ld_b_full(smemW + (size_t)(j * 16 + l15) * 128, half, kk * 32);
    #pragma unroll
    for (int j = 0; j < 4; ++j)
      acc[j] = wmma_bf16(a, bfrag[j], acc[j]);
  }

  int rmask = (1 << n_shift) - 1;
  #pragma unroll
  for (int j = 0; j < 4; ++j) {
    int col = tn4 * 64 + j * 16 + l15;
    float bc = bias[col];
    #pragma unroll
    for (int r = 0; r < 8; ++r) {
      int row = tm * 16 + half * 8 + r;
      float val = acc[j][r] + bc;
      if (mode == 0) {
        if (col < 128)      p0[(size_t)row * 128 + col] = (bf16)(val * qscale);
        else if (col < 256) p1[(size_t)row * 128 + (col - 128)] = (bf16)val;
        else {
          int jj = col - 256, nn = row & rmask, w = row >> n_shift;
          p2[(size_t)(((nn << 3) + (jj >> 4)) * 16 + (jj & 15)) * 128 + w] = (bf16)val;
        }
      } else if (mode == 1) {
        if (col < 128) p0[(size_t)row * 128 + col] = (bf16)val;
        else {
          int jj = col - 128, nn = row & rmask, w = row >> n_shift;
          p2[(size_t)(((nn << 3) + (jj >> 4)) * 16 + (jj & 15)) * 128 + w] = (bf16)val;
        }
      } else if (mode == 2) {
        p0[(size_t)row * 128 + col] = (bf16)(val * qscale);
      } else if (mode == 3) {
        if (col < 128) p0[(size_t)row * 128 + col] = (bf16)(val * qscale);
        else           p1[(size_t)row * 128 + (col - 128)] = (bf16)val;
      } else {
        int nn = row & rmask, w = row >> n_shift;
        featp[(size_t)((w << 7) + n_off + nn) * 128 + col] += val;
      }
    }
  }
}

// ---------------------------------------------------------------------------
// Attention terms (K=16 zero-padded to 32), attn layout [n][e][w][v].
// Each wave: one row-tile x four column-tiles; all B fragments loaded first,
// then 4 WMMAs back-to-back.
// mode 1: batch (n,e): attn[n,e,w,v]  = sum_c q[w,n,e,c] k[v,n,e,c]   (store)
// mode 2: batch (w,e): attn[n,e,w,v] += sum_c q[w,n,e,c] k_r[w,v,e,c] (add)
// mode 3: batch (v,e): attn[n,e,w,v] += sum_c k[v,n,e,c] q_r[w,v,e,c] (add)
// ---------------------------------------------------------------------------
__global__ __launch_bounds__(256)
void k_attn_term(const bf16* __restrict__ Pm, const bf16* __restrict__ Qm,
                 float* __restrict__ attn, int mode, int n_sub, int tx_shift,
                 int batches) {
  int lane = threadIdx.x & 31;
  int g = blockIdx.x * 8 + (threadIdx.x >> 5);
  int ty4 = g & 1;                         // which group of 4 column tiles
  int tx  = (g >> 1) & ((1 << tx_shift) - 1);
  int b   = g >> (1 + tx_shift);
  if (b >= batches) return;
  int half = lane >> 4, l15 = lane & 15;
  int e = b & 7, bidx = b >> 3;
  int m = tx * 16 + l15;                   // A row for this lane
  const bf16* arow;
  if (mode == 1) arow = Pm + (size_t)(m * n_sub + bidx) * 128 + e * 16;
  else           arow = Pm + (size_t)(bidx * n_sub + m) * 128 + e * 16;
  v16bf a = ld_a_k16(arow, half);
  v16bf bfrag[4];
  #pragma unroll
  for (int j = 0; j < 4; ++j) {
    int nc = ty4 * 64 + j * 16 + l15;      // B "N" row for this lane
    const bf16* brow;
    if (mode == 1)      brow = Qm + (size_t)(nc * n_sub + bidx) * 128 + e * 16;
    else if (mode == 2) brow = Qm + (size_t)((bidx << 7) + nc) * 128 + e * 16;
    else                brow = Qm + (size_t)((nc << 7) + bidx) * 128 + e * 16;
    bfrag[j] = ld_b_k16(brow, half);
  }
  v8f acc[4];
  #pragma unroll
  for (int j = 0; j < 4; ++j)
    #pragma unroll
    for (int i = 0; i < 8; ++i) acc[j][i] = 0.0f;
  #pragma unroll
  for (int j = 0; j < 4; ++j)
    acc[j] = wmma_bf16(a, bfrag[j], acc[j]);
  #pragma unroll
  for (int j = 0; j < 4; ++j) {
    int Nc = ty4 * 64 + j * 16 + l15;
    #pragma unroll
    for (int r = 0; r < 8; ++r) {
      int Mr = tx * 16 + half * 8 + r;
      if (mode == 1)
        attn[((size_t)(bidx * 8 + e) << 14) + (Mr << 7) + Nc] = acc[j][r];
      else if (mode == 2)
        attn[(((size_t)(Mr * 8 + e) << 7) + bidx) * 128 + Nc] += acc[j][r];
      else
        attn[(((size_t)(Mr * 8 + e) << 7) + Nc) * 128 + bidx] += acc[j][r];
    }
  }
}

// softmax over last axis (v=128); one wave per (n,e,w) row -> bf16 probs
__global__ __launch_bounds__(256)
void k_softmax(const float* __restrict__ attn, bf16* __restrict__ ab, int rows) {
  int row = blockIdx.x * 8 + (threadIdx.x >> 5);
  if (row >= rows) return;
  int lane = threadIdx.x & 31;
  float4 x = *((const float4*)(attn + (size_t)row * 128) + lane);
  float m = fmaxf(fmaxf(x.x, x.y), fmaxf(x.z, x.w));
  for (int d = 16; d; d >>= 1) m = fmaxf(m, __shfl_xor(m, d, 32));
  float e0 = expf(x.x - m), e1 = expf(x.y - m);
  float e2 = expf(x.z - m), e3 = expf(x.w - m);
  float s = e0 + e1 + e2 + e3;
  for (int d = 16; d; d >>= 1) s += __shfl_xor(s, d, 32);
  float inv = 1.0f / s;
  bf16* dst = ab + (size_t)row * 128 + lane * 4;
  dst[0] = (bf16)(e0 * inv); dst[1] = (bf16)(e1 * inv);
  dst[2] = (bf16)(e2 * inv); dst[3] = (bf16)(e3 * inv);
}

// AV: per (n,e): v_o[w, n, e*16+c] = sum_v a[n,e,w,v] * Vt[n,e,c,v]  (K=128)
// Each wave: four 16-row tiles sharing the B (Vt) operand; per K-step the
// four A fragments are loaded first, then 4 WMMAs back-to-back.
__global__ __launch_bounds__(256)
void k_av(const bf16* __restrict__ ab, const bf16* __restrict__ vt,
          bf16* __restrict__ vo, int n_sub) {
  int lane = threadIdx.x & 31;
  int g = blockIdx.x * 8 + (threadIdx.x >> 5);
  int tw4 = g & 1;                  // which group of 4 row tiles
  int b   = g >> 1;                 // n*8 + e
  if (b >= n_sub * 8) return;
  int n = b >> 3, e = b & 7;
  int half = lane >> 4, l15 = lane & 15;
  const bf16* brow = vt + ((size_t)b * 16 + l15) * 128;
  v8f acc[4];
  #pragma unroll
  for (int j = 0; j < 4; ++j)
    #pragma unroll
    for (int i = 0; i < 8; ++i) acc[j][i] = 0.0f;
  #pragma unroll
  for (int kk = 0; kk < 4; ++kk) {
    v16bf bb = ld_b_full(brow, half, kk * 32);
    v16bf afrag[4];
    #pragma unroll
    for (int j = 0; j < 4; ++j) {
      const bf16* arow = ab + ((size_t)b * 128 + tw4 * 64 + j * 16 + l15) * 128;
      afrag[j] = ld_a_full(arow, half, kk * 32);
    }
    #pragma unroll
    for (int j = 0; j < 4; ++j)
      acc[j] = wmma_bf16(afrag[j], bb, acc[j]);
  }
  #pragma unroll
  for (int j = 0; j < 4; ++j)
    #pragma unroll
    for (int r = 0; r < 8; ++r) {
      int w = tw4 * 64 + j * 16 + half * 8 + r;
      vo[(size_t)(w * n_sub + n) * 128 + e * 16 + l15] = (bf16)acc[j][r];
    }
}

// causal mask on last cross layer: attn[n,e,w,v] = -inf for v > w
__global__ __launch_bounds__(256)
void k_mask(float* __restrict__ attn) {
  int idx = blockIdx.x * 256 + threadIdx.x;
  if (idx >= 64 * 8 * 128 * 128) return;
  int v = idx & 127, w = (idx >> 7) & 127;
  if (v > w) attn[idx] = -__builtin_inff();
}

// raw_attn: out[n,w,v] = sum_e attn[n,e,w,v]   (output (1,64,128,128) f32)
__global__ __launch_bounds__(256)
void k_rawsum(const float* __restrict__ attn, float* __restrict__ out) {
  int idx = blockIdx.x * 256 + threadIdx.x;
  if (idx >= 64 * 128 * 128) return;
  int v = idx & 127, w = (idx >> 7) & 127, n = idx >> 14;
  float s = 0.0f;
  #pragma unroll
  for (int e = 0; e < 8; ++e)
    s += attn[((size_t)(n * 8 + e) << 14) + (w << 7) + v];
  out[idx] = s;
}

// ---------------------------------------------------------------------------
extern "C" void kernel_launch(void* const* d_in, const int* in_sizes, int n_in,
                              void* d_out, int out_size, void* d_ws, size_t ws_size,
                              hipStream_t stream) {
  (void)in_sizes; (void)n_in; (void)out_size; (void)ws_size;
  const float* feat_left   = (const float*)d_in[0];
  const float* feat_right  = (const float*)d_in[1];
  const float* pos_enc     = (const float*)d_in[2];
  const float* self_ln_g   = (const float*)d_in[3];
  const float* self_ln_b   = (const float*)d_in[4];
  const float* self_in_w   = (const float*)d_in[5];
  const float* self_in_b   = (const float*)d_in[6];
  const float* self_out_w  = (const float*)d_in[7];
  const float* self_out_b  = (const float*)d_in[8];
  const float* cross_ln1_g = (const float*)d_in[9];
  const float* cross_ln1_b = (const float*)d_in[10];
  const float* cross_ln2_g = (const float*)d_in[11];
  const float* cross_ln2_b = (const float*)d_in[12];
  const float* cross_in_w  = (const float*)d_in[13];
  const float* cross_in_b  = (const float*)d_in[14];
  const float* cross_out_w = (const float*)d_in[15];
  const float* cross_out_b = (const float*)d_in[16];

  char* p = (char*)d_ws;
  auto carve = [&](size_t bytes) {
    void* r = (void*)p; p += (bytes + 255) & ~(size_t)255; return r;
  };
  float* feat    = (float*)carve((size_t)2097152 * 4);
  bf16* pos_sel  = (bf16*) carve((size_t)2097152 * 2);
  bf16* w_si     = (bf16*) carve((size_t)294912 * 2);   // self_in_w bf16
  bf16* w_ci     = (bf16*) carve((size_t)294912 * 2);   // cross_in_w bf16
  bf16* w_so     = (bf16*) carve((size_t)98304 * 2);
  bf16* w_co     = (bf16*) carve((size_t)98304 * 2);
  bf16* lnb      = (bf16*) carve((size_t)2097152 * 2);
  bf16* qb       = (bf16*) carve((size_t)2097152 * 2);
  bf16* kb       = (bf16*) carve((size_t)2097152 * 2);
  bf16* vtb      = (bf16*) carve((size_t)2097152 * 2);
  bf16* qr       = (bf16*) carve((size_t)2097152 * 2);
  bf16* kr       = (bf16*) carve((size_t)2097152 * 2);
  bf16* vob      = (bf16*) carve((size_t)2097152 * 2);
  float* attn    = (float*)carve((size_t)16777216 * 4);
  bf16* ab       = (bf16*) carve((size_t)16777216 * 2);

  auto nb = [](int n) { return (n + 255) / 256; };

  // setup (re-done every call: deterministic, no cached state)
  k_cvt_bf16<<<nb(294912), 256, 0, stream>>>(self_in_w,  w_si, 294912);
  k_cvt_bf16<<<nb(294912), 256, 0, stream>>>(cross_in_w, w_ci, 294912);
  k_cvt_bf16<<<nb(98304),  256, 0, stream>>>(self_out_w,  w_so, 98304);
  k_cvt_bf16<<<nb(98304),  256, 0, stream>>>(cross_out_w, w_co, 98304);
  k_build_feat<<<nb(2097152), 256, 0, stream>>>(feat_left, feat_right, feat);
  k_build_pos <<<nb(2097152), 256, 0, stream>>>(pos_enc, pos_sel);

  for (int i = 0; i < NLAYERS; ++i) {
    const bf16* Wsi = w_si + (size_t)i * 49152;
    const bf16* Wci = w_ci + (size_t)i * 49152;

    // ===== self attention (n=128, M=16384 -> 1024 M-tiles) =====
    k_layernorm<<<(16384 + 7) / 8, 256, 0, stream>>>(
        feat, self_ln_g + i * 128, self_ln_b + i * 128, lnb, 16384, 7, 0);
    // QKV: M=16384, N=384 -> grid (1024/8)*6
    k_gemm_proj<<<128 * 6, 256, 0, stream>>>(
        lnb, Wsi, self_in_b + i * 384, 16384, 384, 0, 7, 0, 0.25f,
        qb, kb, vtb, nullptr);
    // pos: M=16384, N=256 -> grid (1024/8)*4
    k_gemm_proj<<<128 * 4, 256, 0, stream>>>(
        pos_sel, Wsi, self_in_b + i * 384, 16384, 256, 3, 7, 0, 0.25f,
        qr, kr, nullptr, nullptr);
    // attn terms: waves = batches * tiles_x * 2
    k_attn_term<<<(1024 * 8 * 2) / 8, 256, 0, stream>>>(qb, kb, attn, 1, 128, 3, 1024);
    k_attn_term<<<(1024 * 8 * 2) / 8, 256, 0, stream>>>(qb, kr, attn, 2, 128, 3, 1024);
    k_attn_term<<<(1024 * 8 * 2) / 8, 256, 0, stream>>>(kb, qr, attn, 3, 128, 3, 1024);
    k_softmax<<<131072 / 8, 256, 0, stream>>>(attn, ab, 131072);
    k_av<<<(1024 * 2) / 8, 256, 0, stream>>>(ab, vtb, vob, 128);
    // out proj, add into feat: N=128 -> grid (1024/8)*2
    k_gemm_proj<<<128 * 2, 256, 0, stream>>>(
        vob, w_so + (size_t)i * 16384, self_out_b + i * 128, 16384, 128, 4, 7, 0,
        1.0f, nullptr, nullptr, nullptr, feat);

    // ===== cross attention (n=64, M=8192 -> 512 M-tiles) =====
    k_layernorm<<<8192 / 8, 256, 0, stream>>>(
        feat, cross_ln1_g + i * 128, cross_ln1_b + i * 128, lnb, 8192, 6, 0);
    k_layernorm<<<8192 / 8, 256, 0, stream>>>(
        feat, cross_ln2_g + i * 128, cross_ln2_b + i * 128,
        lnb + (size_t)8192 * 128, 8192, 6, 64);
    // q from left (W rows 0..127): grid (512/8)*2
    k_gemm_proj<<<64 * 2, 256, 0, stream>>>(
        lnb, Wci, cross_in_b + i * 384, 8192, 128, 2, 6, 0, 0.25f,
        qb, nullptr, nullptr, nullptr);
    // k,v from right (W rows 128..383, bias offset 128): grid (512/8)*4
    k_gemm_proj<<<64 * 4, 256, 0, stream>>>(
        lnb + (size_t)8192 * 128, Wci + (size_t)128 * 128,
        cross_in_b + i * 384 + 128, 8192, 256, 1, 6, 0, 1.0f,
        kb, nullptr, vtb, nullptr);
    // pos with cross weights
    k_gemm_proj<<<128 * 4, 256, 0, stream>>>(
        pos_sel, Wci, cross_in_b + i * 384, 16384, 256, 3, 7, 0, 0.25f,
        qr, kr, nullptr, nullptr);
    // attn terms (n_sub=64 -> tx_shift 2 for modes 2/3)
    k_attn_term<<<(512 * 8 * 2) / 8, 256, 0, stream>>>(qb, kb, attn, 1, 64, 3, 512);
    k_attn_term<<<(1024 * 4 * 2) / 8, 256, 0, stream>>>(qb, kr, attn, 2, 64, 2, 1024);
    k_attn_term<<<(1024 * 4 * 2) / 8, 256, 0, stream>>>(kb, qr, attn, 3, 64, 2, 1024);

    if (i == NLAYERS - 1) {
      // causal mask, then raw_attn = sum over heads -> output; done.
      k_mask<<<nb(64 * 8 * 128 * 128), 256, 0, stream>>>(attn);
      k_rawsum<<<nb(64 * 128 * 128), 256, 0, stream>>>(attn, (float*)d_out);
    } else {
      k_softmax<<<65536 / 8, 256, 0, stream>>>(attn, ab, 65536);
      k_av<<<(512 * 2) / 8, 256, 0, stream>>>(ab, vtb, vob, 64);
      // out proj adds into left half of feat (n_off=0, n_sub=64)
      k_gemm_proj<<<64 * 2, 256, 0, stream>>>(
          vob, w_co + (size_t)i * 16384, cross_out_b + i * 128, 8192, 128, 4, 6, 0,
          1.0f, nullptr, nullptr, nullptr, feat);
    }
  }
}
